// MoEAugmentedActor_44212393345184
// MI455X (gfx1250) — compile-verified, matrix operand order checked by reading, not run on hardware
//
#include <hip/hip_runtime.h>
#include <hip/hip_bf16.h>

typedef _Float16 v8h  __attribute__((ext_vector_type(8)));
typedef _Float16 v16h __attribute__((ext_vector_type(16)));
typedef float    v8f  __attribute__((ext_vector_type(8)));

// ---------------- packed-weight layout in d_ws (f16 elements) ----------------
// Tile = 32(K) x 16(N), stored as 32 lanes * 16 halves = 512 f16.
// Within a tile, lane l (l&15 = N col), element j: K = (j<8?0:16) + ((l>=16)?8:0) + (j&7)
// Tile order: ((e*NT + nt)*KT + kt)  -> k-tiles contiguous for a fixed n-tile.
static constexpr int OFF_VW1 = 0;        // vae_W1 480x256 : KT=15 NT=16 -> 122880 (K permuted!)
static constexpr int OFF_VWZ = 122880;   // vae_Wz 256x32  : KT=8  NT=2  -> 8192
static constexpr int OFF_VWV = 131072;   // vae_Wv 256x3   : KT=8  NT=1  -> 4096
static constexpr int OFF_AW1 = 135168;   // ae_W1  96x64   : KT=3  NT=4  -> 6144
static constexpr int OFF_AW2 = 141312;   // ae_W2  64x32   : KT=2  NT=2  -> 2048
static constexpr int OFF_GW1 = 143360;   // gate_W1 32x64  : KT=1  NT=4  -> 2048
static constexpr int OFF_GW2 = 145408;   // gate_W2 64x5   : KT=2  NT=1  -> 1024
static constexpr int OFF_EW1 = 146432;   // eW1 5x166x128  : KT=6  NT=8  -> 122880
static constexpr int OFF_EW2 = 269312;   // eW2 5x128x128  : KT=4  NT=8  -> 81920
static constexpr int OFF_EW3 = 351232;   // eW3 5x128x29   : KT=4  NT=2  -> 20480
// total = 371712 f16 = 743424 bytes of d_ws

// permMode 1: K index k refers to x column (15+k); map back to original
// vae_hist index (frame-major) so the device kernel can stage hist as a
// contiguous copy of x[:, 15:495].
__global__ void pack_w_kernel(const float* __restrict__ W, _Float16* __restrict__ dst,
                              int K, int N, int KT, int NT, int E, int permMode) {
    long total = (long)E * NT * KT * 512;
    for (long i = (long)blockIdx.x * blockDim.x + threadIdx.x; i < total;
         i += (long)gridDim.x * blockDim.x) {
        int  j  = (int)(i & 15);
        int  l  = (int)((i >> 4) & 31);
        long t  = i >> 9;
        int  kt = (int)(t % KT);
        long t2 = t / KT;
        int  nt = (int)(t2 % NT);
        int  e  = (int)(t2 / NT);
        int  k  = kt * 32 + ((j < 8) ? 0 : 16) + ((l >= 16) ? 8 : 0) + (j & 7);
        int  n  = nt * 16 + (l & 15);
        float v = 0.0f;
        if (k < K && n < N) {
            int ks = k;
            if (permMode == 1) {
                int xcol = 15 + k;               // staged hist col k == x col 15+k
                if (xcol < 60) {                  // terms 1..3, dims 3, offsets 15/30/45
                    int w = xcol - 15, tt = w / 15, within = w - tt * 15;
                    int f = within / 3, d = within - f * 3;
                    ks = f * 96 + tt * 3 + d;
                } else {                          // terms 4..6, dims 29, offsets 60/205/350
                    int w = xcol - 60, tt = w / 145, ww = w - tt * 145;
                    int f = ww / 29, d = ww - f * 29;
                    ks = f * 96 + 9 + tt * 29 + d;
                }
            }
            v = W[(long)e * K * N + (long)ks * N + n];
        }
        dst[i] = (_Float16)v;
    }
}

// ---------------- fused MoE actor ----------------
static constexpr int SX = 488;   // halves: hist(480) / inp(192)+h1(128)+h2(128)
static constexpr int SY = 264;   // halves: h(256) / e_t(96)+ha(64)+glh(64)
static constexpr int SZ = 40;    // halves: z_E(32)
static constexpr int PER_WAVE_H = 13952; // bufX 7808 + bufY 4224 + bufZ 640 + G 1024 + W 256

static __device__ __forceinline__ v8f wmma_kloop(const _Float16* __restrict__ aRow,
                                                 const _Float16* __restrict__ bt,
                                                 int nK, v8f c) {
    for (int kk = 0; kk < nK; ++kk) {
        v8h lo = *(const v8h*)(aRow + kk * 32);
        v8h hi = *(const v8h*)(aRow + kk * 32 + 16);
        v16h a = __builtin_shufflevector(lo, hi, 0,1,2,3,4,5,6,7,8,9,10,11,12,13,14,15);
        v16h b = *(const v16h*)(bt + kk * 512);
        c = __builtin_amdgcn_wmma_f32_16x16x32_f16(false, a, false, b, (short)0, c,
                                                   false, false);
    }
    return c;
}

static __device__ __forceinline__ void store_tile(v8f c, _Float16* dst, int strideH,
                                                  int colBase, const float* bias,
                                                  int validN, bool doElu, int lane) {
    int n    = lane & 15;
    int mrow = (lane >> 4) * 8;
    if (n < validN) {
        float b = bias[n];
#pragma unroll
        for (int v = 0; v < 8; ++v) {
            float val = c[v] + b;
            if (doElu) val = val > 0.0f ? val : (__expf(val) - 1.0f);
            dst[(mrow + v) * strideH + colBase + n] = (_Float16)val;
        }
    }
}

__global__ __launch_bounds__(128)
void moe_actor_kernel(const float* __restrict__ x, const _Float16* __restrict__ pk,
                      const float* __restrict__ vae_b1, const float* __restrict__ vae_bz,
                      const float* __restrict__ vae_bv, const float* __restrict__ ae_b1,
                      const float* __restrict__ ae_b2,  const float* __restrict__ gate_b1,
                      const float* __restrict__ gate_b2, const float* __restrict__ eb1,
                      const float* __restrict__ eb2,     const float* __restrict__ eb3,
                      float* __restrict__ out) {
    extern __shared__ _Float16 smem[];
    const int lane = threadIdx.x & 31;
    const int wave = threadIdx.x >> 5;
    const int row0 = (blockIdx.x * 4 + wave) * 16;

    _Float16* S    = smem + wave * PER_WAVE_H;
    _Float16* bufX = S;                       // 16 x SX
    _Float16* bufY = S + 16 * SX;             // 16 x SY
    _Float16* bufZ = S + 16 * SX + 16 * SY;   // 16 x SZ
    float*    bufG = (float*)(bufZ + 16 * SZ);        // 16 x 16 f32 (gating logits)
    float*    bufW = (float*)(bufZ + 16 * SZ + 1024); // 16 x 8  f32 (softmax weights)

    const int m     = lane & 15;
    const int mrow  = (lane >> 4) * 8;
    const int hhalf = (lane >> 4) * 8;        // K offset of this half-wave in A frags
    const _Float16* pkL = pk + lane * 16;     // lane's slice of every B tile

    const _Float16* aX = bufX + m * SX + hhalf;
    const _Float16* aY = bufY + m * SY + hhalf;
    const _Float16* aZ = bufZ + m * SZ + hhalf;

    // ---- stage e_t (x cols 879..974, contiguous) -> bufY[:,0:96]  (coalesced)
    for (int r = 0; r < 16; ++r) {
        const float* xr = x + (long)(row0 + r) * 975;
        for (int c = lane; c < 96; c += 32)
            bufY[r * SY + c] = (_Float16)xr[879 + c];
    }
    __syncthreads();

    // ---- AE L1: ha = elu(e_t @ ae_W1)  -> bufY[:,96:160]
    for (int nt = 0; nt < 4; ++nt) {
        v8f c = {};
        c = wmma_kloop(aY, pkL + OFF_AW1 + nt * 3 * 512, 3, c);
        store_tile(c, bufY, SY, 96 + nt * 16, ae_b1 + nt * 16, 16, true, lane);
    }
    __syncthreads();

    // ---- AE L2: z_E = ha @ ae_W2 -> bufZ[:,0:32]
    for (int nt = 0; nt < 2; ++nt) {
        v8f c = {};
        c = wmma_kloop(aY + 96, pkL + OFF_AW2 + nt * 2 * 512, 2, c);
        store_tile(c, bufZ, SZ, nt * 16, ae_b2 + nt * 16, 16, false, lane);
    }
    __syncthreads();

    // ---- Gate L1: glh = elu(z_E @ gate_W1) -> bufY[:,160:224]
    for (int nt = 0; nt < 4; ++nt) {
        v8f c = {};
        c = wmma_kloop(aZ, pkL + OFF_GW1 + nt * 1 * 512, 1, c);
        store_tile(c, bufY, SY, 160 + nt * 16, gate_b1 + nt * 16, 16, true, lane);
    }
    __syncthreads();

    // ---- Gate L2: gl = glh @ gate_W2 (cols 0..4 valid) -> bufG (f32)
    {
        v8f c = {};
        c = wmma_kloop(aY + 160, pkL + OFF_GW2, 2, c);
        int n = lane & 15;
        float b = (n < 5) ? gate_b2[n] : 0.0f;
#pragma unroll
        for (int v = 0; v < 8; ++v) bufG[(mrow + v) * 16 + n] = c[v] + b;
    }
    __syncthreads();

    // ---- softmax over 5 experts, one row per lane
    if (lane < 16) {
        float g[5], mx = -1e30f;
#pragma unroll
        for (int e = 0; e < 5; ++e) { g[e] = bufG[lane * 16 + e]; mx = fmaxf(mx, g[e]); }
        float s = 0.0f;
#pragma unroll
        for (int e = 0; e < 5; ++e) { g[e] = __expf(g[e] - mx); s += g[e]; }
        float inv = 1.0f / s;
#pragma unroll
        for (int e = 0; e < 5; ++e) bufW[lane * 8 + e] = g[e] * inv;
    }
    __syncthreads();

    // ---- stage vae_hist in x-column order: bufX[:,c] = x[:, 15+c]  (coalesced,
    //      vae_W1 pack kernel applies the matching K-permutation)
    for (int r = 0; r < 16; ++r) {
        const float* xr = x + (long)(row0 + r) * 975;
        for (int c = lane; c < 480; c += 32)
            bufX[r * SX + c] = (_Float16)xr[15 + c];
    }
    __syncthreads();

    // ---- VAE L1: h = elu(hist @ vae_W1) -> bufY[:,0:256]
    for (int nt = 0; nt < 16; ++nt) {
        v8f c = {};
        c = wmma_kloop(aX, pkL + OFF_VW1 + nt * 15 * 512, 15, c);
        store_tile(c, bufY, SY, nt * 16, vae_b1 + nt * 16, 16, true, lane);
    }
    __syncthreads();

    // ---- build inp [o_t(0:99) | v_pred(99:102) | z_H(102:134) | z_E(134:166) | 0(166:192)]
    // z_H = h @ vae_Wz  -> bufX cols 102..133
    for (int nt = 0; nt < 2; ++nt) {
        v8f c = {};
        c = wmma_kloop(aY, pkL + OFF_VWZ + nt * 8 * 512, 8, c);
        store_tile(c, bufX, SX, 102 + nt * 16, vae_bz + nt * 16, 16, false, lane);
    }
    // v_pred = h @ vae_Wv (3 valid cols) -> bufX cols 99..101
    {
        v8f c = {};
        c = wmma_kloop(aY, pkL + OFF_VWV, 8, c);
        store_tile(c, bufX, SX, 99, vae_bv, 3, false, lane);
    }
    // o_t phase 1: gather frame-4 columns into the (currently free) h1 region
    // cols 192..290.  Reads hist cols {12..44,161..189,306..334,451..479} —
    // disjoint from writes to 99..133 and 192..290; cols 161..165 are read before the z_E copy.
    for (int r = 0; r < 16; ++r) {
        const float* xr = x + (long)(row0 + r) * 975;
        for (int j = lane; j < 99; j += 32) {
            _Float16 v;
            if (j < 3) {
                v = (_Float16)xr[12 + j];                 // base_lin_vel frame 4
            } else {
                int xcol;
                if (j < 12) xcol = (j / 3) * 15 + 12 + (j % 3);
                else { int jj = j - 12; xcol = 176 + (jj / 29) * 145 + (jj % 29); }
                v = bufX[r * SX + (xcol - 15)];           // staged hist col
            }
            bufX[r * SX + 192 + j] = v;
        }
    }
    // z_E copy -> bufX cols 134..165
    for (int idx = lane; idx < 16 * 32; idx += 32) {
        int r = idx >> 5, c2 = idx & 31;
        bufX[r * SX + 134 + c2] = bufZ[r * SZ + c2];
    }
    __syncthreads();
    // o_t phase 2: move into inp cols 0..98; zero pad cols 166..191
    for (int idx = lane; idx < 16 * 99; idx += 32) {
        int r = idx / 99, j = idx - r * 99;
        bufX[r * SX + j] = bufX[r * SX + 192 + j];
    }
    for (int idx = lane; idx < 16 * 26; idx += 32) {
        int r = idx / 26, c2 = idx - r * 26;
        bufX[r * SX + 166 + c2] = (_Float16)0.0f;
    }
    __syncthreads();

    // ---- experts: inp(0:192) -> h1(192:320) -> h2(320:448), weighted-accumulate
    v8f acc[2] = {};
    const _Float16* aI  = aX;
    const _Float16* aH1 = aX + 192;
    const _Float16* aH2 = aX + 320;
    for (int e = 0; e < 5; ++e) {
        for (int nt = 0; nt < 8; ++nt) {
            v8f c = {};
            c = wmma_kloop(aI, pkL + OFF_EW1 + (e * 8 + nt) * 6 * 512, 6, c);
            store_tile(c, bufX, SX, 192 + nt * 16, eb1 + e * 128 + nt * 16, 16, true, lane);
        }
        __syncthreads();
        for (int nt = 0; nt < 8; ++nt) {
            v8f c = {};
            c = wmma_kloop(aH1, pkL + OFF_EW2 + (e * 8 + nt) * 4 * 512, 4, c);
            store_tile(c, bufX, SX, 320 + nt * 16, eb2 + e * 128 + nt * 16, 16, true, lane);
        }
        __syncthreads();
        for (int nt = 0; nt < 2; ++nt) {
            v8f c = {};
            c = wmma_kloop(aH2, pkL + OFF_EW3 + (e * 2 + nt) * 4 * 512, 4, c);
            int n = nt * 16 + (lane & 15);
            float bb = (n < 29) ? eb3[e * 29 + n] : 0.0f;
#pragma unroll
            for (int v = 0; v < 8; ++v) {
                float wv = bufW[(mrow + v) * 8 + e];
                acc[nt][v] += wv * (c[v] + bb);
            }
        }
        __syncthreads();
    }

    // ---- store out [N,29] fp32
#pragma unroll
    for (int nt = 0; nt < 2; ++nt) {
        int n = nt * 16 + (lane & 15);
        if (n < 29) {
#pragma unroll
            for (int v = 0; v < 8; ++v)
                out[(long)(row0 + mrow + v) * 29 + n] = acc[nt][v];
        }
    }
}

extern "C" void kernel_launch(void* const* d_in, const int* in_sizes, int n_in,
                              void* d_out, int out_size, void* d_ws, size_t ws_size,
                              hipStream_t stream) {
    (void)in_sizes; (void)n_in; (void)out_size; (void)ws_size;
    const float* x       = (const float*)d_in[0];
    const float* vae_W1  = (const float*)d_in[1];
    const float* vae_b1  = (const float*)d_in[2];
    const float* vae_Wz  = (const float*)d_in[3];
    const float* vae_bz  = (const float*)d_in[4];
    const float* vae_Wv  = (const float*)d_in[5];
    const float* vae_bv  = (const float*)d_in[6];
    const float* ae_W1   = (const float*)d_in[7];
    const float* ae_b1   = (const float*)d_in[8];
    const float* ae_W2   = (const float*)d_in[9];
    const float* ae_b2   = (const float*)d_in[10];
    const float* gate_W1 = (const float*)d_in[11];
    const float* gate_b1 = (const float*)d_in[12];
    const float* gate_W2 = (const float*)d_in[13];
    const float* gate_b2 = (const float*)d_in[14];
    const float* eW1     = (const float*)d_in[15];
    const float* eb1     = (const float*)d_in[16];
    const float* eW2     = (const float*)d_in[17];
    const float* eb2     = (const float*)d_in[18];
    const float* eW3     = (const float*)d_in[19];
    const float* eb3     = (const float*)d_in[20];
    _Float16* pk = (_Float16*)d_ws;

    auto pack = [&](const float* W, int off, int K, int N, int KT, int NT, int E, int pm) {
        long total = (long)E * NT * KT * 512;
        int blocks = (int)((total + 255) / 256);
        pack_w_kernel<<<blocks, 256, 0, stream>>>(W, pk + off, K, N, KT, NT, E, pm);
    };
    pack(vae_W1,  OFF_VW1, 480, 256, 15, 16, 1, 1);   // K permuted to x-column order
    pack(vae_Wz,  OFF_VWZ, 256,  32,  8,  2, 1, 0);
    pack(vae_Wv,  OFF_VWV, 256,   3,  8,  1, 1, 0);
    pack(ae_W1,   OFF_AW1,  96,  64,  3,  4, 1, 0);
    pack(ae_W2,   OFF_AW2,  64,  32,  2,  2, 1, 0);
    pack(gate_W1, OFF_GW1,  32,  64,  1,  4, 1, 0);
    pack(gate_W2, OFF_GW2,  64,   5,  2,  1, 1, 0);
    pack(eW1,     OFF_EW1, 166, 128,  6,  8, 5, 0);
    pack(eW2,     OFF_EW2, 128, 128,  4,  8, 5, 0);
    pack(eW3,     OFF_EW3, 128,  29,  4,  2, 5, 0);

    const int N = 131072;
    const int rowsPerBlock = 64;                 // 4 wave32s x 16 rows
    int grid = N / rowsPerBlock;                 // 2048
    size_t smem = (size_t)4 * PER_WAVE_H * sizeof(_Float16); // ~109 KB (<320 KB/WGP)
    moe_actor_kernel<<<grid, 128, smem, stream>>>(
        x, pk, vae_b1, vae_bz, vae_bv, ae_b1, ae_b2, gate_b1, gate_b2,
        eb1, eb2, eb3, (float*)d_out);
}
